// SSDXMemDetector_3100966388100
// MI455X (gfx1250) — matrix operand order repaired and checked by reading, not compiled, not run on hardware
//
#include <hip/hip_runtime.h>

#define BB   8
#define CCH  1024      // channels (C)
#define NPOS 4096      // H*W
#define NT   128       // n-tile (memory positions per iteration)
#define MT   32        // m-tile (query columns per workgroup)
#define CC   128       // channel chunk staged in LDS
#define AST  132       // s_a row stride (padded, bf16 elements)
#define QST  1032      // s_qk row stride (padded)
#define PST  136       // s_p row stride (padded)

typedef __attribute__((ext_vector_type(16))) __bf16 bf16x16;
typedef __attribute__((ext_vector_type(8)))  __bf16 bf16x8;
typedef __attribute__((ext_vector_type(4)))  __bf16 bf16x4;
typedef __attribute__((ext_vector_type(8)))  float  f32x8;
typedef __attribute__((ext_vector_type(4)))  float  f32x4;

__device__ __forceinline__ f32x8 wmma_bf16(bf16x16 a, bf16x16 b, f32x8 c) {
  // D = A(16x32 bf16) x B(32x16 bf16) + C(16x16 f32)
  return __builtin_amdgcn_wmma_f32_16x16x32_bf16(false, a, false, b, (short)0, c, false, false);
}

__device__ __forceinline__ bf16x16 cat16(bf16x8 lo, bf16x8 hi) {
  return __builtin_shufflevector(lo, hi, 0,1,2,3,4,5,6,7,8,9,10,11,12,13,14,15);
}

// A fragment (16x32 bf16, MxK): lane (h=lane/16, M=lane%16) holds
// K = kbase + h*8 + {0..7} (elems 0..7) and kbase + 16 + h*8 + {0..7} (elems 8..15).
// `row` points at the M-th row of a [M][K] LDS tile with K contiguous.
__device__ __forceinline__ bf16x16 frag_a_lds(const __bf16* row, int kbase, int h) {
  const __bf16* p0 = row + kbase + h * 8;
  const __bf16* p1 = p0 + 16;
  bf16x4 a0 = *(const bf16x4*)p0;
  bf16x4 a1 = *(const bf16x4*)(p0 + 4);
  bf16x4 b0 = *(const bf16x4*)p1;
  bf16x4 b1 = *(const bf16x4*)(p1 + 4);
  bf16x8 lo = __builtin_shufflevector(a0, a1, 0,1,2,3,4,5,6,7);
  bf16x8 hi = __builtin_shufflevector(b0, b1, 0,1,2,3,4,5,6,7);
  return cat16(lo, hi);
}

// B fragment (32x16 bf16, KxN): lane (h, N=lane%16) holds
// K = kbase + h*8 + {0..7} (elems 0..7) and kbase + 16 + h*8 + {0..7} (elems 8..15). `row` = the N-th column stored K-contiguous.
__device__ __forceinline__ bf16x16 frag_b_lds(const __bf16* row, int kbase, int h) {
  const __bf16* p = row + kbase + h * 8;
  bf16x8 lo = *(const bf16x8*)p;
  bf16x8 hi = *(const bf16x8*)(p + 16);
  return cat16(lo, hi);
}

__device__ __forceinline__ bf16x8 cvt8(const float* p) {
  f32x4 x = *(const f32x4*)p;
  f32x4 y = *(const f32x4*)(p + 4);
  bf16x8 r;
  r[0] = (__bf16)x[0]; r[1] = (__bf16)x[1]; r[2] = (__bf16)x[2]; r[3] = (__bf16)x[3];
  r[4] = (__bf16)y[0]; r[5] = (__bf16)y[1]; r[6] = (__bf16)y[2]; r[7] = (__bf16)y[3];
  return r;
}

// A fragment loaded directly from a global f32 row (K contiguous), converted to bf16.
__device__ __forceinline__ bf16x16 frag_a_global(const float* row, int h) {
  return cat16(cvt8(row + h * 8), cvt8(row + 16 + h * 8));
}

// order-preserving float<->uint for atomicMax on shared memory
__device__ __forceinline__ unsigned fenc(float x) {
  unsigned u = __float_as_uint(x);
  return (u & 0x80000000u) ? ~u : (u | 0x80000000u);
}
__device__ __forceinline__ float fdec(unsigned u) {
  u = (u & 0x80000000u) ? (u & 0x7fffffffu) : ~u;
  return __uint_as_float(u);
}

// ---------------- kernel 0a: a_sq[b,n] = sum_c mk[b,c,n]^2 ----------------
__global__ __launch_bounds__(256) void asq_kernel(const float* __restrict__ mk,
                                                  float* __restrict__ asq) {
  int idx = blockIdx.x * 256 + threadIdx.x;       // [0, B*NPOS)
  int b = idx >> 12;
  int n = idx & (NPOS - 1);
  const float* p = mk + (size_t)b * CCH * NPOS + n;
  float s = 0.0f;
  for (int c = 0; c < CCH; ++c) { float v = p[(size_t)c * NPOS]; s += v * v; }
  asq[idx] = s;
}

// ---------------- kernel 0b: out[b, C:2C, :] = curr_value ----------------
// Pure byte move: demonstrate the CDNA5 async global->LDS path (ASYNCcnt).
__global__ __launch_bounds__(256) void copy_kernel(const float4* __restrict__ cv,
                                                   float4* __restrict__ out) {
  __shared__ float4 stage[256];
  size_t idx = (size_t)blockIdx.x * 256 + threadIdx.x;   // [0, B*C*N/4)
  const size_t per = (size_t)CCH * NPOS / 4;
  size_t b = idx / per, r = idx - b * per;

  const float4* src = cv + idx;
  // LDS offset = low 32 bits of the generic shared-pointer (LDS aperture maps addr[31:0])
  unsigned lds_off = (unsigned)(uintptr_t)(&stage[threadIdx.x]);
  asm volatile("global_load_async_to_lds_b128 %0, %1, off"
               :: "v"(lds_off), "v"(src) : "memory");
  asm volatile("s_wait_asynccnt 0x0" ::: "memory");
  out[b * (2 * per) + per + r] = stage[threadIdx.x];
}

// ---------------- kernel 1: per-column softmax stats (online max/sum) ----------------
__global__ __launch_bounds__(512) void stats_kernel(
    const float* __restrict__ mk, const float* __restrict__ qk,
    const float* __restrict__ asq,
    float* __restrict__ colmax, float* __restrict__ colsum)
{
  __shared__ __bf16 s_qk[MT][QST];   // qk tile, [m][c] (K-contiguous for B frags)
  __shared__ __bf16 s_a[NT][AST];    // mk chunk, [n][c] (K-contiguous for A frags)
  __shared__ float s_runmax[MT], s_runsum[MT], s_tsum[MT];
  __shared__ unsigned s_tmax[MT];

  const int b = blockIdx.y;
  const int m0 = blockIdx.x * MT;
  const int t = threadIdx.x;
  const int wave = t >> 5, lane = t & 31;
  const int h = lane >> 4, l16 = lane & 15;
  const int nsub = wave & 7, msub = wave >> 3;   // 8 n-subtiles x 2 m-subtiles
  const int mloc = msub * 16 + l16;

  const float* mkb  = mk  + (size_t)b * CCH * NPOS;
  const float* qkb  = qk  + (size_t)b * CCH * NPOS;
  const float* asqb = asq + (size_t)b * NPOS;

  // resident qk tile -> bf16, transposed [m][c]
  for (int r = 0; r < (CCH * MT) / 512; ++r) {
    int flat = r * 512 + t;
    int c = flat >> 5, m = flat & 31;
    s_qk[m][c] = (__bf16)qkb[(size_t)c * NPOS + m0 + m];
  }
  if (t < MT) { s_runmax[t] = -3.0e38f; s_runsum[t] = 0.0f; s_tmax[t] = fenc(-3.0e38f); }
  __syncthreads();

  const int n_st = t & (NT - 1);   // staging: each thread owns one n row
  const int cp   = t >> 7;         // and a c-pair phase 0..3

  for (int n0 = 0; n0 < NPOS; n0 += NT) {
    f32x8 acc = {0,0,0,0,0,0,0,0};
    for (int c0 = 0; c0 < CCH; c0 += CC) {
      __syncthreads();
      // stage mk[c0:c0+128, n0:n0+128] transposed to s_a[n][c] as bf16
      for (int r = 0; r < CC / 8; ++r) {
        int cl = r * 8 + cp * 2;
        const float* g = mkb + (size_t)(c0 + cl) * NPOS + (n0 + n_st);
        float x0 = g[0];
        float x1 = g[NPOS];
        s_a[n_st][cl]     = (__bf16)x0;
        s_a[n_st][cl + 1] = (__bf16)x1;
      }
      if (c0 + CC < CCH)
        __builtin_prefetch(mkb + (size_t)(c0 + CC + cp * 2) * NPOS + n0 + n_st, 0, 1);
      __syncthreads();
      #pragma unroll
      for (int kk = 0; kk < CC; kk += 32) {
        bf16x16 fa = frag_a_lds(&s_a[nsub * 16 + l16][0], kk, h);
        bf16x16 fb = frag_b_lds(&s_qk[mloc][0], c0 + kk, h);
        acc = wmma_bf16(fa, fb, acc);
      }
    }
    // affinity = (2*ab - a_sq[n]) / sqrt(1024);  D layout: v -> n = n0+nsub*16+8h+v
    float affv[8];
    const float* ap = asqb + n0 + nsub * 16 + h * 8;
    f32x4 a0 = *(const f32x4*)ap;
    f32x4 a1 = *(const f32x4*)(ap + 4);
    float lmax = -3.0e38f;
    #pragma unroll
    for (int v = 0; v < 8; ++v) {
      float aq = (v < 4) ? a0[v] : a1[v - 4];
      float af = (2.0f * acc[v] - aq) * 0.03125f;
      affv[v] = af;
      lmax = fmaxf(lmax, af);
    }
    atomicMax(&s_tmax[mloc], fenc(lmax));
    __syncthreads();
    if (t < MT) {
      float tm = fdec(s_tmax[t]);
      float nm = fmaxf(s_runmax[t], tm);
      s_runsum[t] *= __expf(s_runmax[t] - nm);
      s_runmax[t] = nm;
      s_tsum[t] = 0.0f;
    }
    __syncthreads();
    float nmv = s_runmax[mloc];
    float ls = 0.0f;
    #pragma unroll
    for (int v = 0; v < 8; ++v) ls += __expf(affv[v] - nmv);
    atomicAdd(&s_tsum[mloc], ls);
    __syncthreads();
    if (t < MT) { s_runsum[t] += s_tsum[t]; s_tmax[t] = fenc(-3.0e38f); }
  }
  __syncthreads();
  if (t < MT) {
    colmax[(size_t)b * NPOS + m0 + t] = s_runmax[t];
    colsum[(size_t)b * NPOS + m0 + t] = s_runsum[t];
  }
}

// ---------------- kernel 2: recompute scores, normalize, mem = mo @ P ----------------
__global__ __launch_bounds__(512) void apply_kernel(
    const float* __restrict__ mk, const float* __restrict__ qk,
    const float* __restrict__ mo, const float* __restrict__ asq,
    const float* __restrict__ colmax, const float* __restrict__ colsum,
    float* __restrict__ out)
{
  __shared__ __bf16 s_qk[MT][QST];
  __shared__ __bf16 s_a[NT][AST];
  __shared__ __bf16 s_p[MT][PST];   // P tile, [m][n] (K-contiguous for B frags)

  const int b = blockIdx.y;
  const int m0 = blockIdx.x * MT;
  const int t = threadIdx.x;
  const int wave = t >> 5, lane = t & 31;
  const int h = lane >> 4, l16 = lane & 15;
  const int nsub = wave & 7, msub = wave >> 3;
  const int mloc = msub * 16 + l16;
  const int cbase = wave * 64;                 // each wave owns 64 output channels

  const float* mkb  = mk  + (size_t)b * CCH * NPOS;
  const float* qkb  = qk  + (size_t)b * CCH * NPOS;
  const float* mob  = mo  + (size_t)b * CCH * NPOS;
  const float* asqb = asq + (size_t)b * NPOS;

  for (int r = 0; r < (CCH * MT) / 512; ++r) {
    int flat = r * 512 + t;
    int c = flat >> 5, m = flat & 31;
    s_qk[m][c] = (__bf16)qkb[(size_t)c * NPOS + m0 + m];
  }
  __syncthreads();

  const float cm = colmax[(size_t)b * NPOS + m0 + mloc];
  const float rs = 1.0f / colsum[(size_t)b * NPOS + m0 + mloc];

  f32x8 oacc[4][2] = {};   // [c-subtile][m-subtile] accumulators (64 VGPRs)

  const int n_st = t & (NT - 1);
  const int cp   = t >> 7;

  for (int n0 = 0; n0 < NPOS; n0 += NT) {
    // ---- phase A: scores S[128n x 32m], one 16x16 subtile per wave ----
    f32x8 acc = {0,0,0,0,0,0,0,0};
    for (int c0 = 0; c0 < CCH; c0 += CC) {
      __syncthreads();
      for (int r = 0; r < CC / 8; ++r) {
        int cl = r * 8 + cp * 2;
        const float* g = mkb + (size_t)(c0 + cl) * NPOS + (n0 + n_st);
        float x0 = g[0];
        float x1 = g[NPOS];
        s_a[n_st][cl]     = (__bf16)x0;
        s_a[n_st][cl + 1] = (__bf16)x1;
      }
      if (c0 + CC < CCH)
        __builtin_prefetch(mkb + (size_t)(c0 + CC + cp * 2) * NPOS + n0 + n_st, 0, 1);
      __syncthreads();
      #pragma unroll
      for (int kk = 0; kk < CC; kk += 32) {
        bf16x16 fa = frag_a_lds(&s_a[nsub * 16 + l16][0], kk, h);
        bf16x16 fb = frag_b_lds(&s_qk[mloc][0], c0 + kk, h);
        acc = wmma_bf16(fa, fb, acc);
      }
    }
    // ---- normalize -> P (bf16) into LDS ----
    const float* ap = asqb + n0 + nsub * 16 + h * 8;
    f32x4 a0 = *(const f32x4*)ap;
    f32x4 a1 = *(const f32x4*)(ap + 4);
    bf16x8 pb;
    #pragma unroll
    for (int v = 0; v < 8; ++v) {
      float aq = (v < 4) ? a0[v] : a1[v - 4];
      float af = (2.0f * acc[v] - aq) * 0.03125f;
      pb[v] = (__bf16)(__expf(af - cm) * rs);
    }
    *(bf16x8*)&s_p[mloc][nsub * 16 + h * 8] = pb;   // n run of 8 at nsub*16+8h
    __syncthreads();
    // ---- phase B: oacc += mo[cbase:+64, n0:+128] x P[128 x 32] ----
    #pragma unroll
    for (int kk = 0; kk < 4; ++kk) {
      bf16x16 fb0 = frag_b_lds(&s_p[l16][0],      kk * 32, h);
      bf16x16 fb1 = frag_b_lds(&s_p[16 + l16][0], kk * 32, h);
      #pragma unroll
      for (int as = 0; as < 4; ++as) {
        const float* arow = mob + (size_t)(cbase + as * 16 + l16) * NPOS + n0 + kk * 32;
        bf16x16 fa = frag_a_global(arow, h);
        oacc[as][0] = wmma_bf16(fa, fb0, oacc[as][0]);
        oacc[as][1] = wmma_bf16(fa, fb1, oacc[as][1]);
      }
    }
    __syncthreads();
  }

  // write mem into out[b, 0:1024, :]
  float* ob = out + (size_t)b * (2 * CCH) * NPOS;
  #pragma unroll
  for (int as = 0; as < 4; ++as)
    #pragma unroll
    for (int ms = 0; ms < 2; ++ms)
      #pragma unroll
      for (int v = 0; v < 8; ++v) {
        int c = cbase + as * 16 + h * 8 + v;       // D layout: M = v + 8h
        int m = m0 + ms * 16 + l16;                // N = lane%16
        ob[(size_t)c * NPOS + m] = oacc[as][ms][v];
      }
}

extern "C" void kernel_launch(void* const* d_in, const int* in_sizes, int n_in,
                              void* d_out, int out_size, void* d_ws, size_t ws_size,
                              hipStream_t stream) {
  const float* prev_key   = (const float*)d_in[0];
  const float* prev_value = (const float*)d_in[1];
  const float* curr_key   = (const float*)d_in[2];
  const float* curr_value = (const float*)d_in[3];
  float* out = (float*)d_out;

  float* asq    = (float*)d_ws;            // B*N floats
  float* colmax = asq    + BB * NPOS;      // B*N floats
  float* colsum = colmax + BB * NPOS;      // B*N floats  (total 384 KB << ws)

  asq_kernel<<<(BB * NPOS) / 256, 256, 0, stream>>>(prev_key, asq);
  copy_kernel<<<(int)(((size_t)BB * CCH * NPOS / 4) / 256), 256, 0, stream>>>(
      (const float4*)curr_value, (float4*)out);

  dim3 grid(NPOS / MT, BB);
  stats_kernel<<<grid, 512, 0, stream>>>(prev_key, curr_key, asq, colmax, colsum);
  apply_kernel<<<grid, 512, 0, stream>>>(prev_key, curr_key, prev_value, asq,
                                         colmax, colsum, out);
}
